// TiedAxialSelfAttention2d_86620900425797
// MI455X (gfx1250) — matrix operand order repaired and checked by reading, not compiled, not run on hardware
//
#include <hip/hip_runtime.h>
#include <hip/hip_bf16.h>

// ---------------------------------------------------------------------------
// TiedAxialSelfAttention2d on gfx1250 (MI455X), wave32 + WMMA bf16.
// B=1, E=128, L=256, H=12, DH=64, D=768, F=3D=2304, M=E*L=32768.
// All GEMMs via v_wmma_f32_16x16x32_bf16 (bf16 operands, f32 accum).
// QKV GEMM double-buffers its LDS tiles with global_load_async_to_lds_b128
// (ASYNCcnt-tracked async copies, s_wait_asynccnt for pipelining).
// ---------------------------------------------------------------------------

typedef __bf16 bf16;
typedef __attribute__((ext_vector_type(16))) __bf16 v16bf;
typedef __attribute__((ext_vector_type(8)))  __bf16 v8bf;
typedef __attribute__((ext_vector_type(8)))  float  v8f;

#define E_DIM   128
#define L_DIM   256
#define H_DIM   12
#define DH_DIM  64
#define D_DIM   768
#define F_DIM   2304
#define M_ROWS  32768

// ---- WMMA wrapper ----------------------------------------------------------
__device__ inline v8f wmma_bf16(v16bf a, v16bf b, v8f c) {
  // (neg_a, A, neg_b, B, c_mod, C, reuse_a, reuse_b)
  return __builtin_amdgcn_wmma_f32_16x16x32_bf16(false, a, false, b,
                                                 (short)0, c, false, false);
}

// ---- async global->LDS copy of 16B (CDNA5 ASYNCcnt path) -------------------
// Flat LDS addresses carry the LDS byte offset in addr[31:0] (ISA 10.2),
// so truncating the generic pointer yields the per-lane LDS address VGPR.
__device__ inline void async_ld_b128(void* lds, const void* gptr) {
  unsigned l = (unsigned)(uintptr_t)lds;
  unsigned long long g = (unsigned long long)(uintptr_t)gptr;
  asm volatile("global_load_async_to_lds_b128 %0, %1, off"
               :: "v"(l), "v"(g) : "memory");
}

__device__ inline void wait_async_le4() {
#if __has_builtin(__builtin_amdgcn_s_wait_asynccnt)
  __builtin_amdgcn_s_wait_asynccnt(4);
#else
  asm volatile("s_wait_asynccnt 0x4" ::: "memory");
#endif
}
__device__ inline void wait_async_0() {
#if __has_builtin(__builtin_amdgcn_s_wait_asynccnt)
  __builtin_amdgcn_s_wait_asynccnt(0);
#else
  asm volatile("s_wait_asynccnt 0x0" ::: "memory");
#endif
}

// ---- A fragment: 16x32 bf16, row-major source, K contiguous ---------------
// lane<16 : row=lane,    K = {0..7} U {16..23}
// lane>=16: row=lane-16, K = {8..15} U {24..31}
__device__ inline v16bf frag_a(const bf16* base, size_t stride) {
  int lane = threadIdx.x & 31;
  const bf16* p = base + (size_t)(lane & 15) * stride + (size_t)((lane >> 4) * 8);
  v8bf lo = *(const v8bf*)(p);
  v8bf hi = *(const v8bf*)(p + 16);
  v16bf r;
#pragma unroll
  for (int i = 0; i < 8; ++i) { r[i] = lo[i]; r[8 + i] = hi[i]; }
  return r;
}

// ---- B fragment: 32x16 bf16 given B^T row-major (N rows, K contiguous) ----
// lane<16 : col=lane,    K = {0..7} U {16..23}
// lane>=16: col=lane-16, K = {8..15} U {24..31}
__device__ inline v16bf frag_b(const bf16* baseT, size_t stride) {
  int lane = threadIdx.x & 31;
  const bf16* p = baseT + (size_t)(lane & 15) * stride + (size_t)((lane >> 4) * 8);
  v8bf lo = *(const v8bf*)(p);
  v8bf hi = *(const v8bf*)(p + 16);
  v16bf r;
#pragma unroll
  for (int i = 0; i < 8; ++i) { r[i] = lo[i]; r[8 + i] = hi[i]; }
  return r;
}

// ---- C/D fragment element mapping: VGPR r -> row (lane>>4)*8+r, col lane&15
#define CFRAG_ROW(lane, r) (((lane) >> 4) * 8 + (r))
#define CFRAG_COL(lane)    ((lane) & 15)

// ===========================================================================
// fp32 -> bf16 convert (weights)
// ===========================================================================
__global__ void k_cvt_bf16(const float* __restrict__ src, bf16* __restrict__ dst, int n) {
  int i = blockIdx.x * blockDim.x + threadIdx.x;
  if (i < n) dst[i] = (bf16)src[i];
}

// ===========================================================================
// LayerNorm over D=768 per row, emit bf16. 256 threads/row, 3 elems/thread.
// ===========================================================================
__global__ __launch_bounds__(256) void k_ln_bf16(const float* __restrict__ x,
                                                 const float* __restrict__ g,
                                                 const float* __restrict__ b,
                                                 bf16* __restrict__ out) {
  int row = blockIdx.x;
  int t = threadIdx.x;
  const float* xr = x + (size_t)row * D_DIM;
  float v0 = xr[t], v1 = xr[t + 256], v2 = xr[t + 512];
  __shared__ float red[256];
  red[t] = v0 + v1 + v2;
  __syncthreads();
#pragma unroll
  for (int off = 128; off > 0; off >>= 1) {
    if (t < off) red[t] += red[t + off];
    __syncthreads();
  }
  float mu = red[0] * (1.0f / D_DIM);
  __syncthreads();
  float d0 = v0 - mu, d1 = v1 - mu, d2 = v2 - mu;
  red[t] = d0 * d0 + d1 * d1 + d2 * d2;
  __syncthreads();
#pragma unroll
  for (int off = 128; off > 0; off >>= 1) {
    if (t < off) red[t] += red[t + off];
    __syncthreads();
  }
  float rs = rsqrtf(red[0] * (1.0f / D_DIM) + 1e-5f);
  bf16* o = out + (size_t)row * D_DIM;
  o[t]       = (bf16)(d0 * rs * g[t]       + b[t]);
  o[t + 256] = (bf16)(d1 * rs * g[t + 256] + b[t + 256]);
  o[t + 512] = (bf16)(d2 * rs * g[t + 512] + b[t + 512]);
}

// ===========================================================================
// QKV GEMM: C[32768,2304] = A[32768,768] x W^T (+bias), bf16 in/out, f32 acc.
// 256 thr = 8 waves as 2(M)x4(N); wave tile 64x32; block tile 128x128.
// Operand tiles double-buffered in LDS via async global->LDS copies.
// ===========================================================================
__global__ __launch_bounds__(256) void k_gemm_qkv(const bf16* __restrict__ A,
                                                  const bf16* __restrict__ W,  // [2304,768] = B^T
                                                  const float* __restrict__ bias,
                                                  bf16* __restrict__ C) {
  int bm = blockIdx.x * 128, bn = blockIdx.y * 128;
  int wave = threadIdx.x >> 5;
  int wmL = (wave >> 2) * 64;   // block-local M of this wave
  int wnL = (wave & 3) * 32;    // block-local N of this wave

  __shared__ bf16 sA[2][128][32];   // 2 x 8KB
  __shared__ bf16 sW[2][128][32];   // 2 x 8KB

  int t = threadIdx.x;
  int row0 = t >> 2;             // 0..63
  int c0   = (t & 3) * 8;        // 0,8,16,24

  // stage one 128x32 K-slab of A and W into LDS buffer `buf` (4 async / thread)
  auto stage = [&](int buf, int k0) {
    async_ld_b128(&sA[buf][row0][c0],      A + (size_t)(bm + row0) * D_DIM + k0 + c0);
    async_ld_b128(&sA[buf][row0 + 64][c0], A + (size_t)(bm + row0 + 64) * D_DIM + k0 + c0);
    async_ld_b128(&sW[buf][row0][c0],      W + (size_t)(bn + row0) * D_DIM + k0 + c0);
    async_ld_b128(&sW[buf][row0 + 64][c0], W + (size_t)(bn + row0 + 64) * D_DIM + k0 + c0);
  };

  v8f acc[4][2];
#pragma unroll
  for (int mi = 0; mi < 4; ++mi)
#pragma unroll
    for (int ni = 0; ni < 2; ++ni)
#pragma unroll
      for (int r = 0; r < 8; ++r) acc[mi][ni][r] = 0.0f;

  const int NK = D_DIM / 32;   // 24
  stage(0, 0);
  for (int it = 0; it < NK; ++it) {
    if (it + 1 < NK) {
      stage((it + 1) & 1, (it + 1) * 32);   // overlap next copy with this compute
      wait_async_le4();                     // previous stage (4/thread) complete
    } else {
      wait_async_0();
    }
    __syncthreads();

    const bf16 (*cA)[32] = sA[it & 1];
    const bf16 (*cW)[32] = sW[it & 1];
    v16bf bf[2];
#pragma unroll
    for (int ni = 0; ni < 2; ++ni)
      bf[ni] = frag_b(&cW[wnL + ni * 16][0], 32);
#pragma unroll
    for (int mi = 0; mi < 4; ++mi) {
      v16bf af = frag_a(&cA[wmL + mi * 16][0], 32);
#pragma unroll
      for (int ni = 0; ni < 2; ++ni)
        acc[mi][ni] = wmma_bf16(af, bf[ni], acc[mi][ni]);
    }
    __syncthreads();   // everyone done reading before this buffer is re-filled
  }

  int lane = threadIdx.x & 31;
#pragma unroll
  for (int mi = 0; mi < 4; ++mi)
#pragma unroll
    for (int ni = 0; ni < 2; ++ni)
#pragma unroll
      for (int r = 0; r < 8; ++r) {
        size_t row = (size_t)(bm + wmL + mi * 16 + CFRAG_ROW(lane, r));
        int col = bn + wnL + ni * 16 + CFRAG_COL(lane);
        C[row * F_DIM + col] = (bf16)(acc[mi][ni][r] + bias[col]);
      }
}

// ===========================================================================
// rowmask[h][j] = sum_e attn_mask[h][e][j]
// ===========================================================================
__global__ __launch_bounds__(256) void k_rowmask(const float* __restrict__ mask,
                                                 float* __restrict__ rmask) {
  int h = blockIdx.x, j = threadIdx.x;
  float s = 0.f;
  for (int e = 0; e < E_DIM; ++e)
    s += mask[((size_t)(h * E_DIM) + e) * L_DIM + j];
  rmask[h * L_DIM + j] = s;
}

// ===========================================================================
// Row scores (tied over s,c; K = E*DH = 8192) + softmax -> bf16 maps.
// grid (4 i-tiles of 64, 12 heads), 512 thr = 16 waves as 2(M)x8(N).
// ===========================================================================
__global__ __launch_bounds__(512) void k_row_scores(const bf16* __restrict__ proj,
                                                    const float* __restrict__ rmask,
                                                    bf16* __restrict__ rmaps) {
  int it = blockIdx.x;   // i-tile (64 rows)
  int h  = blockIdx.y;
  __shared__ float sm[64][256];  // 64KB
  int wave = threadIdx.x >> 5;
  int wm = (wave >> 3) * 32;   // local M in [0,64)
  int wn = (wave & 7) * 32;    // N in [0,256)
  v8f acc[2][2];
#pragma unroll
  for (int mi = 0; mi < 2; ++mi)
#pragma unroll
    for (int ni = 0; ni < 2; ++ni)
#pragma unroll
      for (int r = 0; r < 8; ++r) acc[mi][ni][r] = 0.0f;

  for (int kk = 0; kk < 256; ++kk) {   // K = 8192 in steps of 32
    int s = kk >> 1;
    int coff = (kk & 1) * 32;
    const bf16* qb = proj + (size_t)(s * L_DIM + it * 64) * F_DIM + h * DH_DIM + coff;
    const bf16* kb = proj + (size_t)(s * L_DIM) * F_DIM + D_DIM + h * DH_DIM + coff;
    v16bf bf[2];
#pragma unroll
    for (int ni = 0; ni < 2; ++ni)
      bf[ni] = frag_b(kb + (size_t)(wn + ni * 16) * F_DIM, F_DIM);
#pragma unroll
    for (int mi = 0; mi < 2; ++mi) {
      v16bf af = frag_a(qb + (size_t)(wm + mi * 16) * F_DIM, F_DIM);
#pragma unroll
      for (int ni = 0; ni < 2; ++ni)
        acc[mi][ni] = wmma_bf16(af, bf[ni], acc[mi][ni]);
    }
  }
  int lane = threadIdx.x & 31;
#pragma unroll
  for (int mi = 0; mi < 2; ++mi)
#pragma unroll
    for (int ni = 0; ni < 2; ++ni)
#pragma unroll
      for (int r = 0; r < 8; ++r)
        sm[wm + mi * 16 + CFRAG_ROW(lane, r)][wn + ni * 16 + CFRAG_COL(lane)] = acc[mi][ni][r];
  __syncthreads();

  int t = threadIdx.x;
  if (t < 64) {
    float mx = -1e30f;
    for (int j = 0; j < 256; ++j) {
      float v = sm[t][j] + rmask[h * L_DIM + j];
      sm[t][j] = v;
      mx = fmaxf(mx, v);
    }
    float ssum = 0.f;
    for (int j = 0; j < 256; ++j) {
      float e = __expf(sm[t][j] - mx);
      ssum += e;
      sm[t][j] = e;
    }
    float inv = 1.0f / ssum;
    bf16* mrow = rmaps + ((size_t)h * L_DIM + it * 64 + t) * L_DIM;
    for (int j = 0; j < 256; ++j) mrow[j] = (bf16)(sm[t][j] * inv);
  }
}

// ===========================================================================
// Row out + residual: out[s,i,h,c] = x + sum_j maps[h,i,j] * v[s,j,h,c].
// grid (s=128, h=12), 256 thr = 8 waves as 4(M=256 i)x2(N=64 c); wave 64x32.
// V^T staged through LDS for the B operand.
// ===========================================================================
__global__ __launch_bounds__(256) void k_row_out(const bf16* __restrict__ proj,
                                                 const bf16* __restrict__ rmaps,
                                                 const float* __restrict__ x,
                                                 float* __restrict__ xo) {
  int s = blockIdx.x, h = blockIdx.y;
  __shared__ bf16 vt[64][48];  // [c][j-local], padded stride (96B, 16B-aligned)
  int wave = threadIdx.x >> 5;
  int wm = (wave >> 1) * 64;   // i
  int wn = (wave & 1) * 32;    // c
  v8f acc[4][2];
#pragma unroll
  for (int mi = 0; mi < 4; ++mi)
#pragma unroll
    for (int ni = 0; ni < 2; ++ni)
#pragma unroll
      for (int r = 0; r < 8; ++r) acc[mi][ni][r] = 0.0f;

  for (int j0 = 0; j0 < 256; j0 += 32) {
    int t = threadIdx.x;
    int jl = t >> 3;               // 0..31
    int c0 = (t & 7) * 8;          // 0..56
    v8bf val = *(const v8bf*)(proj + (size_t)(s * L_DIM + j0 + jl) * F_DIM +
                              2 * D_DIM + h * DH_DIM + c0);
    __syncthreads();  // previous iteration done reading vt
#pragma unroll
    for (int u = 0; u < 8; ++u) vt[c0 + u][jl] = val[u];
    __syncthreads();

    v16bf bf[2];
#pragma unroll
    for (int ni = 0; ni < 2; ++ni)
      bf[ni] = frag_b(&vt[wn + ni * 16][0], 48);
#pragma unroll
    for (int mi = 0; mi < 4; ++mi) {
      v16bf af = frag_a(rmaps + ((size_t)h * L_DIM + wm + mi * 16) * L_DIM + j0, L_DIM);
#pragma unroll
      for (int ni = 0; ni < 2; ++ni)
        acc[mi][ni] = wmma_bf16(af, bf[ni], acc[mi][ni]);
    }
  }
  int lane = threadIdx.x & 31;
#pragma unroll
  for (int mi = 0; mi < 4; ++mi)
#pragma unroll
    for (int ni = 0; ni < 2; ++ni)
#pragma unroll
      for (int r = 0; r < 8; ++r) {
        int i = wm + mi * 16 + CFRAG_ROW(lane, r);
        int c = wn + ni * 16 + CFRAG_COL(lane);
        size_t idx = ((size_t)(s * L_DIM + i)) * D_DIM + h * DH_DIM + c;
        xo[idx] = x[idx] + acc[mi][ni][r];
      }
}

// ===========================================================================
// Column scores per (h,l): [E,E] = Q K^T over c (K=64), + mask, softmax over j.
// grid (l=256), 256 thr = 8 waves as 4(Mi)x2(Nj); wave 32x64.
// ===========================================================================
__global__ __launch_bounds__(256) void k_col_scores(const bf16* __restrict__ proj,
                                                    const float* __restrict__ mask,
                                                    bf16* __restrict__ cmaps,
                                                    int h) {
  int l = blockIdx.x;
  __shared__ float sm[128][128];  // 64KB
  int wave = threadIdx.x >> 5;
  int wm = (wave >> 1) * 32;   // i
  int wn = (wave & 1) * 64;    // j
  const size_t rstr = (size_t)L_DIM * F_DIM;
  v8f acc[2][4];
#pragma unroll
  for (int mi = 0; mi < 2; ++mi)
#pragma unroll
    for (int ni = 0; ni < 4; ++ni)
#pragma unroll
      for (int r = 0; r < 8; ++r) acc[mi][ni][r] = 0.0f;

  for (int k0 = 0; k0 < DH_DIM; k0 += 32) {
    v16bf bf[4];
#pragma unroll
    for (int ni = 0; ni < 4; ++ni)
      bf[ni] = frag_b(proj + (size_t)(wn + ni * 16) * rstr + (size_t)l * F_DIM +
                      D_DIM + h * DH_DIM + k0, rstr);
#pragma unroll
    for (int mi = 0; mi < 2; ++mi) {
      v16bf af = frag_a(proj + (size_t)(wm + mi * 16) * rstr + (size_t)l * F_DIM +
                        h * DH_DIM + k0, rstr);
#pragma unroll
      for (int ni = 0; ni < 4; ++ni)
        acc[mi][ni] = wmma_bf16(af, bf[ni], acc[mi][ni]);
    }
  }
  int lane = threadIdx.x & 31;
#pragma unroll
  for (int mi = 0; mi < 2; ++mi)
#pragma unroll
    for (int ni = 0; ni < 4; ++ni)
#pragma unroll
      for (int r = 0; r < 8; ++r)
        sm[wm + mi * 16 + CFRAG_ROW(lane, r)][wn + ni * 16 + CFRAG_COL(lane)] = acc[mi][ni][r];
  __syncthreads();

  int t = threadIdx.x;
  if (t < 128) {
    float mx = -1e30f;
    for (int j = 0; j < 128; ++j) {
      float v = sm[t][j] + mask[((size_t)(h * E_DIM) + j) * L_DIM + l];
      sm[t][j] = v;
      mx = fmaxf(mx, v);
    }
    float ssum = 0.f;
    for (int j = 0; j < 128; ++j) {
      float e = __expf(sm[t][j] - mx);
      ssum += e;
      sm[t][j] = e;
    }
    float inv = 1.0f / ssum;
    bf16* mrow = cmaps + ((size_t)l * E_DIM + t) * E_DIM;
    for (int j = 0; j < 128; ++j) mrow[j] = (bf16)(sm[t][j] * inv);
  }
}

// ===========================================================================
// Column out + residual into d_out: out[i,l,h,c] += sum_j P[i,j] v[j,l,h,c].
// grid (l=256), 256 thr = 8 waves as 4(Mi=128)x2(Nc=64); wave 32x32.
// ===========================================================================
__global__ __launch_bounds__(256) void k_col_out(const bf16* __restrict__ proj,
                                                 const bf16* __restrict__ cmaps,
                                                 float* __restrict__ xo,
                                                 int h) {
  int l = blockIdx.x;
  __shared__ bf16 vt[64][48];
  int wave = threadIdx.x >> 5;
  int wm = (wave >> 1) * 32;   // i
  int wn = (wave & 1) * 32;    // c
  const size_t rstr = (size_t)L_DIM * F_DIM;
  v8f acc[2][2];
#pragma unroll
  for (int mi = 0; mi < 2; ++mi)
#pragma unroll
    for (int ni = 0; ni < 2; ++ni)
#pragma unroll
      for (int r = 0; r < 8; ++r) acc[mi][ni][r] = 0.0f;

  for (int j0 = 0; j0 < 128; j0 += 32) {
    int t = threadIdx.x;
    int jl = t >> 3;
    int c0 = (t & 7) * 8;
    v8bf val = *(const v8bf*)(proj + (size_t)(j0 + jl) * rstr + (size_t)l * F_DIM +
                              2 * D_DIM + h * DH_DIM + c0);
    __syncthreads();
#pragma unroll
    for (int u = 0; u < 8; ++u) vt[c0 + u][jl] = val[u];
    __syncthreads();

    v16bf bf[2];
#pragma unroll
    for (int ni = 0; ni < 2; ++ni)
      bf[ni] = frag_b(&vt[wn + ni * 16][0], 48);
#pragma unroll
    for (int mi = 0; mi < 2; ++mi) {
      v16bf af = frag_a(cmaps + ((size_t)l * E_DIM + wm + mi * 16) * E_DIM + j0, E_DIM);
#pragma unroll
      for (int ni = 0; ni < 2; ++ni)
        acc[mi][ni] = wmma_bf16(af, bf[ni], acc[mi][ni]);
    }
  }
  int lane = threadIdx.x & 31;
#pragma unroll
  for (int mi = 0; mi < 2; ++mi)
#pragma unroll
    for (int ni = 0; ni < 2; ++ni)
#pragma unroll
      for (int r = 0; r < 8; ++r) {
        int i = wm + mi * 16 + CFRAG_ROW(lane, r);
        int c = wn + ni * 16 + CFRAG_COL(lane);
        size_t idx = ((size_t)(i * L_DIM + l)) * D_DIM + h * DH_DIM + c;
        xo[idx] = xo[idx] + acc[mi][ni][r];
      }
}

// ===========================================================================
extern "C" void kernel_launch(void* const* d_in, const int* in_sizes, int n_in,
                              void* d_out, int out_size, void* d_ws, size_t ws_size,
                              hipStream_t stream) {
  const float* x     = (const float*)d_in[0];
  const float* mask  = (const float*)d_in[1];
  const float* w_row = (const float*)d_in[2];
  const float* b_row = (const float*)d_in[3];
  const float* w_col = (const float*)d_in[4];
  const float* b_col = (const float*)d_in[5];
  const float* g1    = (const float*)d_in[6];
  const float* be1   = (const float*)d_in[7];
  const float* g2    = (const float*)d_in[8];
  const float* be2   = (const float*)d_in[9];
  float* out = (float*)d_out;

  char* ws = (char*)d_ws;
  size_t off = 0;
  bf16* h_bf   = (bf16*)(ws + off); off += (size_t)M_ROWS * D_DIM * 2;       // 48 MB
  bf16* w_bf   = (bf16*)(ws + off); off += (size_t)F_DIM * D_DIM * 2;        // 3.4 MB
  bf16* proj   = (bf16*)(ws + off); off += (size_t)M_ROWS * F_DIM * 2;       // 151 MB
  float* rmask = (float*)(ws + off); off += (size_t)H_DIM * L_DIM * 4;       // 12 KB
  bf16* rmaps  = (bf16*)(ws + off); off += (size_t)H_DIM * L_DIM * L_DIM * 2;// 1.5 MB
  bf16* cmaps  = (bf16*)(ws + off); off += (size_t)L_DIM * E_DIM * E_DIM * 2;// 8.4 MB

  const int nW = F_DIM * D_DIM;

  // -------- pass 1: tied row attention --------
  k_cvt_bf16<<<(nW + 255) / 256, 256, 0, stream>>>(w_row, w_bf, nW);
  k_ln_bf16 <<<M_ROWS, 256, 0, stream>>>(x, g1, be1, h_bf);
  k_gemm_qkv<<<dim3(M_ROWS / 128, F_DIM / 128), 256, 0, stream>>>(h_bf, w_bf, b_row, proj);
  k_rowmask <<<H_DIM, 256, 0, stream>>>(mask, rmask);
  k_row_scores<<<dim3(L_DIM / 64, H_DIM), 512, 0, stream>>>(proj, rmask, rmaps);
  k_row_out <<<dim3(E_DIM, H_DIM), 256, 0, stream>>>(proj, rmaps, x, out);  // out = x + row_out

  // -------- pass 2: tied column attention --------
  k_cvt_bf16<<<(nW + 255) / 256, 256, 0, stream>>>(w_col, w_bf, nW);
  k_ln_bf16 <<<M_ROWS, 256, 0, stream>>>(out, g2, be2, h_bf);
  k_gemm_qkv<<<dim3(M_ROWS / 128, F_DIM / 128), 256, 0, stream>>>(h_bf, w_bf, b_col, proj);
  for (int h = 0; h < H_DIM; ++h) {
    k_col_scores<<<L_DIM, 256, 0, stream>>>(proj, mask, cmaps, h);
    k_col_out   <<<L_DIM, 256, 0, stream>>>(proj, cmaps, out, h);  // out += col_out
  }
}
